// RollerCmp_70480413327831
// MI455X (gfx1250) — matrix operand order repaired and checked by reading, not compiled, not run on hardware
//
#include <hip/hip_runtime.h>
#include <cmath>

// ---------------------------------------------------------------------------
// CDNA5 / gfx1250, wave32.
//  - Stage 1/2 tiny per-head MLPs: V_WMMA_F32_16X16X32_BF16 with K zero-padded
//    16->32 (8192 MAC/instr) -> pipeline is HBM-bound (~524 MB @ 23.3 TB/s).
//  - Final projection GEMM: V_WMMA_F32_16X16X4_F32 (full fp32, K=1024).
//  - Activation: branch-free tanh-form GELU (1 v_exp + 1 v_rcp).
//  - Lane reduction via DS_SWIZZLE_B32 xor-butterfly (immediate lane pattern).
// ---------------------------------------------------------------------------

typedef float  v2f   __attribute__((ext_vector_type(2)));
typedef float  v8f   __attribute__((ext_vector_type(8)));
typedef __bf16 v16bf __attribute__((ext_vector_type(16)));

// D(16x16,f32) = A(16x4,f32) x B(4x16,f32) + C
__device__ __forceinline__ v8f wmma_f32(v2f a, v2f b, v8f c) {
  return __builtin_amdgcn_wmma_f32_16x16x4_f32(false, a, false, b, (short)0, c,
                                               false, false);
}

// D(16x16,f32) = A(16x32,bf16) x B(32x16,bf16) + C
__device__ __forceinline__ v8f wmma_bf16(v16bf a, v16bf b, v8f c) {
  return __builtin_amdgcn_wmma_f32_16x16x32_bf16(false, a, false, b, (short)0, c,
                                                 false, false);
}

// ds_swizzle_b32 xor-butterfly add (group-of-32: and=0x1f, or=0, xor=MASK).
template <int MASK>
__device__ __forceinline__ float swz_xor_add(float s) {
  return s + __int_as_float(__builtin_amdgcn_ds_swizzle(
                 __float_as_int(s), (MASK << 10) | 0x1f));
}

// gelu(x) ~= x * sigmoid(1.595769122x + 0.071354816x^3); branch-free.
__device__ __forceinline__ float gelu_fast(float x) {
  const float s = x * x;
  const float z = x * __builtin_fmaf(0.071354816f, s, 1.595769122f);
  const float e = __expf(-z);                       // v_mul + v_exp_f32
  return x * __builtin_amdgcn_rcpf(1.0f + e);       // v_add + v_rcp_f32 + v_mul
}

// ---------------------------------------------------------------------------
// One block handles a 16(K=CR rows) x 512(d) slab; 256 threads = 8 waves;
// wave w owns head slice d in [w*64, w*64+64).
// H1(64dh x 64f) = X^T(64x16) * W1(16x64): per (mt,ft) ONE bf16 WMMA (K padded
// to 32 with zeros); all 4 ft WMMAs issued back-to-back so the 16-bit
// WMMA->VALU RAW hazard (4 co-exec NOPs) hides under the other WMMAs.
// Then out[dh] = sum_f gelu(H1+b1) * W2[f] + b2 via ds_swizzle reduction
// (C/D layout: VGPR j holds M=j|8+j per half, N = lane%16).
// ---------------------------------------------------------------------------
template <bool ADD_PE>
__device__ __forceinline__ void head_mlp_block(
    const float* __restrict__ src, long long rowStride,
    const float* __restrict__ pe,          // 16 x 512 slice, row stride 1024
    const float* __restrict__ W1,          // 8 heads x (16x64)
    const float* __restrict__ b1,          // 8 heads x 64
    const float* __restrict__ W2,          // 8 heads x 64
    const float* __restrict__ b2,          // 8 heads
    float* __restrict__ dst)               // 512-float output slice
{
  __shared__ float Xs[16][520];            // pad 520 -> conflict-free column reads
  const int tid = threadIdx.x;

  // Stage 16x512 floats (32 KB) into LDS, fusing the +dim_pe add.
  for (int i = tid; i < 2048; i += 256) {  // 2048 float4
    const int row = i >> 7;                // 128 float4 per row
    const int col = (i & 127) << 2;
    float4 v = *(const float4*)(src + (long long)row * rowStride + col);
    if (ADD_PE) {
      const float4 p = *(const float4*)(pe + row * 1024 + col);
      v.x += p.x; v.y += p.y; v.z += p.z; v.w += p.w;
    }
    *(float4*)&Xs[row][col] = v;
  }
  __syncthreads();

  const int wave = tid >> 5;
  const int lane = tid & 31;
  const int lm   = lane & 15;              // N / M index inside a 16-tile
  const int lh   = lane >> 4;              // half of the wave

  const float* W1h = W1 + wave * (16 * 64);
  const float* b1h = b1 + wave * 64;
  const float* W2h = W2 + wave * 64;
  const float  b2v = b2[wave];

  float b1v[4], w2v[4];                    // per-lane scalars: f = ft*16 + lm
#pragma unroll
  for (int ft = 0; ft < 4; ++ft) {
    b1v[ft] = b1h[ft * 16 + lm];
    w2v[ft] = W2h[ft * 16 + lm];
  }

  // B = W1 (32k x 16f tile, K 16..31 zero).  16-bit B layout (wave32):
  // lanes 0-15: elements 0-7 <-> K=0-7; lanes 16-31: elements 0-7 <-> K=8-15;
  // elements 8-15 <-> K=16-31 -> zero padding.
  v16bf bw[4];
#pragma unroll
  for (int ft = 0; ft < 4; ++ft) {
    const int f = ft * 16 + lm;
#pragma unroll
    for (int e = 0; e < 16; ++e)
      bw[ft][e] = (e < 8) ? (__bf16)W1h[(lh * 8 + e) * 64 + f] : (__bf16)0.0f;
  }

#pragma unroll
  for (int mt = 0; mt < 4; ++mt) {         // dh tiles of 16
    const int dloc = wave * 64 + mt * 16 + lm;
    // A = X^T (16m x 32k tile).  16-bit A layout (wave32):
    // lanes 0-15: elements 0-7 <-> K=0-7; lanes 16-31: elements 0-7 <-> K=8-15;
    // elements 8-15 <-> K=16-31 -> zero padding.
    v16bf a;
#pragma unroll
    for (int e = 0; e < 8; ++e) a[e] = (__bf16)Xs[lh * 8 + e][dloc];
#pragma unroll
    for (int e = 8; e < 16; ++e) a[e] = (__bf16)0.0f;

    // Issue all 4 hidden-tile WMMAs first (hazard hiding), consume after.
    v8f cacc[4];
#pragma unroll
    for (int ft = 0; ft < 4; ++ft) {
      v8f z;
#pragma unroll
      for (int j = 0; j < 8; ++j) z[j] = 0.0f;
      cacc[ft] = wmma_bf16(a, bw[ft], z);  // K=16 (padded 32) in one WMMA
    }

    float partial[8];
#pragma unroll
    for (int j = 0; j < 8; ++j) partial[j] = 0.0f;
#pragma unroll
    for (int ft = 0; ft < 4; ++ft)
#pragma unroll
      for (int j = 0; j < 8; ++j)
        partial[j] = __builtin_fmaf(gelu_fast(cacc[ft][j] + b1v[ft]),
                                    w2v[ft], partial[j]);

    // Reduce over N (16 lanes of each half) with ds_swizzle xor-butterfly,
    // then 2 lanes write per j (lane j -> M=j, lane 16+j -> M=8+j).
#pragma unroll
    for (int j = 0; j < 8; ++j) {
      float s = partial[j];
      s = swz_xor_add<1>(s);
      s = swz_xor_add<2>(s);
      s = swz_xor_add<4>(s);
      s = swz_xor_add<8>(s);
      if (lm == j) dst[wave * 64 + mt * 16 + lh * 8 + j] = s + b2v;
    }
  }
}

// ---- stage 1: v-MLP over CR=16 L-rows; grid = B*T*CL*2 = 16000 ----
__global__ __launch_bounds__(256)
void v_stage_kernel(const float* __restrict__ mem, const float* __restrict__ pe,
                    const float* __restrict__ W1, const float* __restrict__ b1,
                    const float* __restrict__ W2, const float* __restrict__ b2,
                    float* __restrict__ vout)
{
  int i = blockIdx.x;
  const int dhalf = i & 1;  i >>= 1;
  const int cl    = i & 15; i >>= 4;
  const int t     = i % 250;
  const int b     = i / 250;
  // x[b, l=cl*16+k, t, :]  ->  mem[b, l*250 + t, :], row stride = 250*1024
  const float* src = mem + ((long long)b * 64001 + (long long)(cl * 16) * 250 + t) * 1024
                         + dhalf * 512;
  float* dst = vout + (((long long)(b * 16 + cl)) * 256 + t) * 1024 + dhalf * 512;
  head_mlp_block<true>(src, 250LL * 1024, pe + dhalf * 512,
                       W1 + dhalf * 8 * 16 * 64, b1 + dhalf * 8 * 64,
                       W2 + dhalf * 8 * 64,      b2 + dhalf * 8, dst);
}

// ---- fill padded rows t in [250,256) with mem[b, L*T, :] ("nonsense") ----
__global__ __launch_bounds__(256)
void pad_kernel(const float* __restrict__ mem, float* __restrict__ vout)
{
  const int i = blockIdx.x * 256 + threadIdx.x;   // 2*16*6*1024 = 196608
  if (i >= 196608) return;
  const int d   = i & 1023;
  const int r   = i >> 10;                        // b*96 + cl*6 + (t-250)
  const int b   = r / 96;
  const int rem = r % 96;
  const int cl  = rem / 6;
  const int t   = 250 + rem % 6;
  vout[(((long long)(b * 16 + cl)) * 256 + t) * 1024 + d] =
      mem[((long long)b * 64001 + 64000) * 1024 + d];
}

// ---- stage 2: h-MLP over T-groups of 16; grid = B*CL*16*2 = 1024 ----
__global__ __launch_bounds__(256)
void h_stage_kernel(const float* __restrict__ vout,
                    const float* __restrict__ W1, const float* __restrict__ b1,
                    const float* __restrict__ W2, const float* __restrict__ b2,
                    float* __restrict__ hbuf)
{
  int i = blockIdx.x;
  const int dhalf = i & 1;  i >>= 1;
  const int tg    = i & 15; i >>= 4;
  const int cl    = i & 15; i >>= 4;
  const int b     = i;
  const float* src = vout + (((long long)(b * 16 + cl)) * 256 + tg * 16) * 1024
                          + dhalf * 512;
  float* dst = hbuf + ((long long)(b * 256 + cl * 16 + tg)) * 1024 + dhalf * 512;
  head_mlp_block<false>(src, 1024, nullptr,
                        W1 + dhalf * 8 * 16 * 64, b1 + dhalf * 8 * 64,
                        W2 + dhalf * 8 * 64,      b2 + dhalf * 8, dst);
}

// ---- stage 3: out(512x1024) = H(512x1024) @ projW^T + projb, f32 WMMA ----
__global__ __launch_bounds__(256)
void proj_kernel(const float* __restrict__ H, const float* __restrict__ Wp,
                 const float* __restrict__ bp, float* __restrict__ out)
{
  __shared__ float As[64][68];
  __shared__ float Bs[64][68];
  const int tid  = threadIdx.x;
  const int r0   = blockIdx.x * 64;          // M
  const int n0   = blockIdx.y * 64;          // N
  const int wave = tid >> 5, lane = tid & 31, lm = lane & 15, lh = lane >> 4;
  const int mt     = wave & 3;
  const int ntBase = (wave >> 2) * 2;

  v8f acc[2];
#pragma unroll
  for (int t = 0; t < 2; ++t)
#pragma unroll
    for (int j = 0; j < 8; ++j) acc[t][j] = 0.0f;

  for (int kb = 0; kb < 1024; kb += 64) {
    for (int i = tid; i < 1024; i += 256) {  // 64x64 floats each, as float4
      const int row = i >> 4;
      const int col = (i & 15) << 2;
      *(float4*)&As[row][col] = *(const float4*)(H  + (long long)(r0 + row) * 1024 + kb + col);
      *(float4*)&Bs[row][col] = *(const float4*)(Wp + (long long)(n0 + row) * 1024 + kb + col);
    }
    if (kb + 64 < 1024) {                    // global_prefetch_b8 of next K-slab
      __builtin_prefetch(H  + (long long)(r0 + (tid >> 2)) * 1024 + kb + 64, 0, 1);
      __builtin_prefetch(Wp + (long long)(n0 + (tid >> 2)) * 1024 + kb + 64, 0, 1);
    }
    __syncthreads();
#pragma unroll
    for (int t = 0; t < 2; ++t) {
      const int nt = ntBase + t;
#pragma unroll
      for (int kc = 0; kc < 16; ++kc) {
        const int k0 = kc * 4 + lh * 2;
        v2f a, b;
        a.x = As[mt * 16 + lm][k0 + 0];
        a.y = As[mt * 16 + lm][k0 + 1];
        b.x = Bs[nt * 16 + lm][k0 + 0];      // Bs holds projW row n -> B[k,n]
        b.y = Bs[nt * 16 + lm][k0 + 1];
        acc[t] = wmma_f32(a, b, acc[t]);
      }
    }
    __syncthreads();
  }

#pragma unroll
  for (int t = 0; t < 2; ++t) {
    const int n  = n0 + (ntBase + t) * 16 + lm;
    const float pb = bp[n];
#pragma unroll
    for (int j = 0; j < 8; ++j) {
      const int r = r0 + mt * 16 + lh * 8 + j;
      out[(long long)r * 1024 + n] = acc[t][j] + pb;
    }
  }
}

// ---------------------------------------------------------------------------
extern "C" void kernel_launch(void* const* d_in, const int* in_sizes, int n_in,
                              void* d_out, int out_size, void* d_ws, size_t ws_size,
                              hipStream_t stream) {
  (void)in_sizes; (void)n_in; (void)out_size; (void)ws_size;
  const float* mem   = (const float*)d_in[0];
  // d_in[1] = lm_emb: unused by the reference (shape source only)
  const float* pe    = (const float*)d_in[2];
  const float* vW1   = (const float*)d_in[3];
  const float* vb1   = (const float*)d_in[4];
  const float* vW2   = (const float*)d_in[5];
  const float* vb2   = (const float*)d_in[6];
  const float* hW1   = (const float*)d_in[7];
  const float* hb1   = (const float*)d_in[8];
  const float* hW2   = (const float*)d_in[9];
  const float* hb2   = (const float*)d_in[10];
  const float* projW = (const float*)d_in[11];
  const float* projb = (const float*)d_in[12];

  float* vout = (float*)d_ws;                               // (B,CL,256,D) = 32 MB
  float* hbuf = vout + (size_t)2 * 16 * 256 * 1024;         // (B,256,D)    =  2 MB
  float* out  = (float*)d_out;                              // (B,256,OUT)

  pad_kernel    <<<768,   256, 0, stream>>>(mem, vout);
  v_stage_kernel<<<16000, 256, 0, stream>>>(mem, pe, vW1, vb1, vW2, vb2, vout);
  h_stage_kernel<<<1024,  256, 0, stream>>>(vout, hW1, hb1, hW2, hb2, hbuf);
  dim3 g3(8, 16);
  proj_kernel   <<<g3,    256, 0, stream>>>(hbuf, projW, projb, out);
}
